// Attention_59373627899977
// MI455X (gfx1250) — matrix operand order repaired and checked by reading, not compiled, not run on hardware
//
#include <hip/hip_runtime.h>
#include <stdint.h>

#define DMODEL 4096
#define BATCH  2
#define SEQ    2048
#define NHEADS 32
#define HDIM   128
#define MTOT   4096   // BATCH*SEQ

typedef unsigned short bf16_t;                       // raw bf16 bits
typedef __attribute__((ext_vector_type(16))) __bf16 v16bf;
typedef __attribute__((ext_vector_type(8)))  float  v8f;
typedef __attribute__((ext_vector_type(4)))  unsigned int v4u;
typedef __attribute__((ext_vector_type(8)))  int v8i;
typedef __attribute__((ext_vector_type(4)))  int v4i;

__device__ __forceinline__ bf16_t f32_to_bf16(float f) {
  __bf16 h = (__bf16)f;                              // hardware cvt
  return __builtin_bit_cast(bf16_t, h);
}

union Frag { v16bf v; uint4 q[2]; bf16_t h[16]; };
union Acc  { v8f v; float f[8]; };

// ---------------------------------------------------------------------------
// TDM: 2-D tile load (global -> LDS), bf16 elements (data_size=1 -> 2B).
// dims/strides in elements. pad_interval code: 2^(c+1) DWORDs between pads;
// pad_amount code: (c+1) DWORDs inserted.
// ---------------------------------------------------------------------------
__device__ __forceinline__ void tdm_load_2d(
    unsigned lds_off, const void* gaddr,
    unsigned tile_d0, unsigned tile_d1,
    unsigned tensor_d0, unsigned tensor_d1,
    unsigned stride_elems, unsigned pad_interval_c, unsigned pad_amount_c,
    unsigned pad_en)
{
  unsigned long long ga = (unsigned long long)(uintptr_t)gaddr;
  v4u g0;
  g0[0] = 1u;                                        // count=1, user descriptor
  g0[1] = lds_off;                                   // lds_addr (bytes)
  g0[2] = (unsigned)(ga & 0xFFFFFFFFu);              // global_addr[31:0]
  g0[3] = (unsigned)((ga >> 32) & 0x1FFFFFFu) | (2u << 30);  // [56:32] | type=2
  v8i g1;
  g1[0] = (int)((1u << 16) |                         // data_size=1 (2 bytes)
                (pad_en << 20) |
                ((pad_interval_c & 7u) << 22) |
                ((pad_amount_c & 127u) << 25));
  g1[1] = (int)((tensor_d0 & 0xFFFFu) << 16);        // tensor_dim0[15:0] @bit48
  g1[2] = (int)(((tensor_d0 >> 16) & 0xFFFFu) |      // tensor_dim0[31:16]
                ((tensor_d1 & 0xFFFFu) << 16));      // tensor_dim1[15:0]
  g1[3] = (int)(((tensor_d1 >> 16) & 0xFFFFu) |      // tensor_dim1[31:16]
                ((tile_d0 & 0xFFFFu) << 16));        // tile_dim0
  g1[4] = (int)(tile_d1 & 0xFFFFu);                  // tile_dim1 (tile_dim2=0)
  g1[5] = (int)stride_elems;                         // tensor_dim0_stride[31:0]
  g1[6] = 0;                                         // stride hi, dim1_stride lo
  g1[7] = 0;
  v4i z4 = {0, 0, 0, 0};
  v8i z8 = {0, 0, 0, 0, 0, 0, 0, 0};
  __builtin_amdgcn_tensor_load_to_lds(g0, g1, z4, z4, z8, 0);
}

// ---------------------------------------------------------------------------
// NT GEMM: C[m,n] = sum_k A[m,k] * B[n,k];  M=K=4096, N=4096.
// Block: 256 threads (8 waves), tile 128(M) x 256(N); wave tile 64x64 (4x4).
// Double-buffered LDS software pipeline:
//   iter i: WMMA on buf[i%2] | VGPR->LDS stores of step i+1 into buf[(i+1)%2]
//           | global loads of step i+2 into VGPRs | ONE barrier.
// A_IS_F32: 1 -> A f32 (cvt to bf16 during staging), 0 -> A bf16.
// CMODE: 0 = bf16 [m,n] | 1 = bf16 V-transposed [b,h,d,s] | 2 = f32 [m,n]
// ---------------------------------------------------------------------------
template <int A_IS_F32, int CMODE>
__global__ __launch_bounds__(256) void gemm_nt_wmma(
    const void* __restrict__ Ap, const float* __restrict__ Bp,
    void* __restrict__ Cp)
{
  __shared__ __align__(16) bf16_t As[2][128][40];   // 20KB
  __shared__ __align__(16) bf16_t Bs[2][256][40];   // 40KB

  const int tid  = threadIdx.x;
  const int lane = tid & 31;
  const int wave = tid >> 5;
  const int wm   = wave >> 2;          // 0..1
  const int wn   = wave & 3;           // 0..3
  const int mblk = (int)blockIdx.y * 128;
  const int nblk = (int)blockIdx.x * 256;
  const int half = lane >> 4;
  const int l16  = lane & 15;

  Acc acc[4][4];
  #pragma unroll
  for (int mi = 0; mi < 4; ++mi)
    #pragma unroll
    for (int ni = 0; ni < 4; ++ni)
      #pragma unroll
      for (int j = 0; j < 8; ++j) acc[mi][ni].f[j] = 0.0f;

  const int ldrow = tid >> 1;          // 0..127
  const int ldcol = (tid & 1) * 16;    // 0 | 16

  // register staging for the software pipeline
  float4 gaf[4];  uint4 ga16[2];  float4 gbf[2][4];

  auto loadA = [&](int k0) {
    if (A_IS_F32) {
      const float4* src = (const float4*)((const float*)Ap +
                          (long)(mblk + ldrow) * DMODEL + k0 + ldcol);
      #pragma unroll
      for (int v = 0; v < 4; ++v) gaf[v] = src[v];
    } else {
      const uint4* src = (const uint4*)((const bf16_t*)Ap +
                          (long)(mblk + ldrow) * DMODEL + k0 + ldcol);
      ga16[0] = src[0];
      ga16[1] = src[1];
    }
  };
  auto loadB = [&](int k0) {
    #pragma unroll
    for (int p = 0; p < 2; ++p) {
      const float4* src = (const float4*)(Bp +
                          (long)(nblk + ldrow + p * 128) * DMODEL + k0 + ldcol);
      #pragma unroll
      for (int v = 0; v < 4; ++v) gbf[p][v] = src[v];
    }
  };
  auto stageA = [&](int buf) {
    if (A_IS_F32) {
      #pragma unroll
      for (int v = 0; v < 4; ++v) {
        As[buf][ldrow][ldcol + v * 4 + 0] = f32_to_bf16(gaf[v].x);
        As[buf][ldrow][ldcol + v * 4 + 1] = f32_to_bf16(gaf[v].y);
        As[buf][ldrow][ldcol + v * 4 + 2] = f32_to_bf16(gaf[v].z);
        As[buf][ldrow][ldcol + v * 4 + 3] = f32_to_bf16(gaf[v].w);
      }
    } else {
      *(uint4*)&As[buf][ldrow][ldcol]     = ga16[0];
      *(uint4*)&As[buf][ldrow][ldcol + 8] = ga16[1];
    }
  };
  auto stageB = [&](int buf) {
    #pragma unroll
    for (int p = 0; p < 2; ++p) {
      const int r = ldrow + p * 128;
      #pragma unroll
      for (int v = 0; v < 4; ++v) {
        Bs[buf][r][ldcol + v * 4 + 0] = f32_to_bf16(gbf[p][v].x);
        Bs[buf][r][ldcol + v * 4 + 1] = f32_to_bf16(gbf[p][v].y);
        Bs[buf][r][ldcol + v * 4 + 2] = f32_to_bf16(gbf[p][v].z);
        Bs[buf][r][ldcol + v * 4 + 3] = f32_to_bf16(gbf[p][v].w);
      }
    }
  };

  const int NSTEP = DMODEL / 32;       // 128
  // prologue: stage k=0, prefetch k=32 into registers
  loadA(0);  loadB(0);
  stageA(0); stageB(0);
  loadA(32); loadB(32);
  __syncthreads();

  for (int i = 0; i < NSTEP; ++i) {
    const int cur = i & 1;
    const int nxt = cur ^ 1;

    // fragments for step i from buf `cur`
    Frag a[4];
    #pragma unroll
    for (int mi = 0; mi < 4; ++mi) {
      const int r  = wm * 64 + mi * 16 + l16;
      const int cb = half * 8;
      a[mi].q[0] = *(const uint4*)&As[cur][r][cb];
      a[mi].q[1] = *(const uint4*)&As[cur][r][cb + 16];
    }

    // stage step i+1 into `nxt` (safe: all waves finished reading nxt at i-1)
    if (i + 1 < NSTEP) { stageA(nxt); stageB(nxt); }
    // prefetch step i+2 into registers
    if (i + 2 < NSTEP) { loadA((i + 2) * 32); loadB((i + 2) * 32); }

    // compute: stream B fragments, A resident
    #pragma unroll
    for (int ni = 0; ni < 4; ++ni) {
      Frag b;
      const int r  = wn * 64 + ni * 16 + l16;
      const int kb = half * 8;
      b.q[0] = *(const uint4*)&Bs[cur][r][kb];
      b.q[1] = *(const uint4*)&Bs[cur][r][kb + 16];
      #pragma unroll
      for (int mi = 0; mi < 4; ++mi)
        acc[mi][ni].v = __builtin_amdgcn_wmma_f32_16x16x32_bf16(
            false, a[mi].v, false, b.v, (short)0, acc[mi][ni].v, false, false);
    }

    __syncthreads();   // nxt fully staged; cur free for overwrite next iter
  }

  #pragma unroll
  for (int mi = 0; mi < 4; ++mi) {
    #pragma unroll
    for (int ni = 0; ni < 4; ++ni) {
      #pragma unroll
      for (int r = 0; r < 8; ++r) {
        const int row = mblk + wm * 64 + mi * 16 + r + half * 8;
        const int col = nblk + wn * 64 + ni * 16 + l16;
        const float val = acc[mi][ni].f[r];
        if (CMODE == 2) {
          ((float*)Cp)[(long)row * DMODEL + col] = val;
        } else if (CMODE == 0) {
          ((bf16_t*)Cp)[(long)row * DMODEL + col] = f32_to_bf16(val);
        } else {  // V transposed: [b,h,d,s]
          const int bb = row >> 11, ss = row & 2047;
          const int hh = col >> 7,  dd = col & 127;
          ((bf16_t*)Cp)[((long)(bb * NHEADS + hh) * HDIM + dd) * SEQ + ss] = f32_to_bf16(val);
        }
      }
    }
  }
}

// ---------------------------------------------------------------------------
// Flash attention (causal). One block = 8 waves = one 128-row Q band of one
// (b,h). K (32x128) and V^T (128x32) tiles are TDM-DMA'd into double-buffered
// LDS (padded by the TDM for conflict-free fragment reads) and shared by all
// 8 waves. Block-uniform trip count; per-wave causal predication.
// ---------------------------------------------------------------------------
__global__ __launch_bounds__(256) void flash_attn_wmma(
    const bf16_t* __restrict__ Qb, const bf16_t* __restrict__ Kb,
    const bf16_t* __restrict__ Vt, bf16_t* __restrict__ Ob)
{
  __shared__ __align__(16) bf16_t Ks[2][32][136];   // 272B row stride
  __shared__ __align__(16) bf16_t Vs[2][128][40];   // 80B row stride
  __shared__ __align__(16) bf16_t Ps[8][16][40];    // per-wave P scratch

  const int tid   = threadIdx.x;
  const int lane  = tid & 31;
  const int wave  = tid >> 5;
  const int qband = (int)blockIdx.x & 15;
  const int h     = ((int)blockIdx.x >> 4) & 31;
  const int b     = (int)blockIdx.x >> 9;
  const int q0    = qband * 128 + wave * 16;
  const int half  = lane >> 4;
  const int l16   = lane & 15;
  const float scale = 0.088388347648318447f;       // 1/sqrt(128)
  const int nIter = qband * 4 + 4;                 // (qband*128+128)/32

  const bf16_t* Kbase = Kb + (long)b * SEQ * DMODEL + h * HDIM;
  const bf16_t* Vbase = Vt + (long)(b * NHEADS + h) * HDIM * SEQ;

  // Resident Q fragments (16 x 128)
  Frag qf[4];
  {
    const bf16_t* qptr = Qb + (long)(b * SEQ + q0 + l16) * DMODEL + h * HDIM;
    #pragma unroll
    for (int dc = 0; dc < 4; ++dc) {
      const int cb = dc * 32 + half * 8;
      qf[dc].q[0] = *(const uint4*)(qptr + cb);
      qf[dc].q[1] = *(const uint4*)(qptr + cb + 16);
    }
  }

  Acc acc[8];
  #pragma unroll
  for (int i = 0; i < 8; ++i)
    #pragma unroll
    for (int j = 0; j < 8; ++j) acc[i].f[j] = 0.0f;
  float mrow[8], lrow[8];
  #pragma unroll
  for (int r = 0; r < 8; ++r) { mrow[r] = -1e30f; lrow[r] = 0.0f; }

  unsigned kOff[2], vOff[2];
  kOff[0] = (unsigned)(uintptr_t)&Ks[0][0][0];
  kOff[1] = (unsigned)(uintptr_t)&Ks[1][0][0];
  vOff[0] = (unsigned)(uintptr_t)&Vs[0][0][0];
  vOff[1] = (unsigned)(uintptr_t)&Vs[1][0][0];

  // Prologue: TDM prefetch of first K/V tiles (wave0: K, wave1: V)
  if (wave == 0) tdm_load_2d(kOff[0], Kbase, 128, 32, 128, 32, DMODEL, 5, 3, 1);
  if (wave == 1) tdm_load_2d(vOff[0], Vbase, 32, 128, 32, 128, SEQ,    3, 3, 1);

  for (int i = 0; i < nIter; ++i) {
    const int kb  = i * 32;
    const int cur = i & 1;
    const int nxt = cur ^ 1;

    if (wave < 2) __builtin_amdgcn_s_wait_tensorcnt(0);
    __syncthreads();                                // tile `cur` ready for all

    if (i + 1 < nIter) {                            // prefetch next tile
      const int kn = kb + 32;
      if (wave == 0)
        tdm_load_2d(kOff[nxt], Kbase + (long)kn * DMODEL, 128, 32, 128, 32, DMODEL, 5, 3, 1);
      if (wave == 1)
        tdm_load_2d(vOff[nxt], Vbase + kn, 32, 128, 32, 128, SEQ, 3, 3, 1);
    }

    if (kb < q0 + 16) {                             // causal predication
      // ---- scores: two 16x16 tiles from LDS K ----
      Acc s[2];
      #pragma unroll
      for (int nt = 0; nt < 2; ++nt)
        #pragma unroll
        for (int j = 0; j < 8; ++j) s[nt].f[j] = 0.0f;

      #pragma unroll
      for (int nt = 0; nt < 2; ++nt) {
        const int krow = nt * 16 + l16;
        #pragma unroll
        for (int dc = 0; dc < 4; ++dc) {
          Frag kf;
          const int cb = dc * 32 + half * 8;
          kf.q[0] = *(const uint4*)&Ks[cur][krow][cb];
          kf.q[1] = *(const uint4*)&Ks[cur][krow][cb + 16];
          s[nt].v = __builtin_amdgcn_wmma_f32_16x16x32_bf16(
              false, qf[dc].v, false, kf.v, (short)0, s[nt].v, false, false);
        }
      }

      // ---- online softmax ----
      float pr[2][8];
      #pragma unroll
      for (int r = 0; r < 8; ++r) {
        const int qrow = q0 + r + half * 8;
        float mx = -1e30f;
        #pragma unroll
        for (int nt = 0; nt < 2; ++nt) {
          const int col = kb + nt * 16 + l16;
          float v = s[nt].f[r] * scale;
          if (col > qrow) v = -1e30f;
          s[nt].f[r] = v;
          mx = fmaxf(mx, v);
        }
        #pragma unroll
        for (int off = 1; off < 16; off <<= 1)
          mx = fmaxf(mx, __shfl_xor(mx, off, 32));
        const float mn = fmaxf(mrow[r], mx);
        const float alpha = __expf(mrow[r] - mn);
        mrow[r] = mn;
        float rs = 0.0f;
        #pragma unroll
        for (int nt = 0; nt < 2; ++nt) {
          const float p = __expf(s[nt].f[r] - mn);
          pr[nt][r] = p;
          rs += p;
        }
        #pragma unroll
        for (int off = 1; off < 16; off <<= 1)
          rs += __shfl_xor(rs, off, 32);
        lrow[r] = lrow[r] * alpha + rs;
        #pragma unroll
        for (int i2 = 0; i2 < 8; ++i2) acc[i2].f[r] *= alpha;
      }

      // ---- P: C-layout -> A-layout via per-wave LDS ----
      #pragma unroll
      for (int nt = 0; nt < 2; ++nt)
        #pragma unroll
        for (int r = 0; r < 8; ++r)
          Ps[wave][r + half * 8][nt * 16 + l16] = f32_to_bf16(pr[nt][r]);
      asm volatile("s_wait_dscnt 0x0" ::: "memory");

      Frag pf;
      {
        const int cb = half * 8;
        pf.q[0] = *(const uint4*)&Ps[wave][l16][cb];
        pf.q[1] = *(const uint4*)&Ps[wave][l16][cb + 16];
      }
      asm volatile("" ::: "memory");

      // ---- PV from LDS V^T ----
      #pragma unroll
      for (int nt = 0; nt < 8; ++nt) {
        const int dcol = nt * 16 + l16;
        Frag vf;
        vf.q[0] = *(const uint4*)&Vs[cur][dcol][half * 8];
        vf.q[1] = *(const uint4*)&Vs[cur][dcol][half * 8 + 16];
        acc[nt].v = __builtin_amdgcn_wmma_f32_16x16x32_bf16(
            false, pf.v, false, vf.v, (short)0, acc[nt].v, false, false);
      }
    }
  }

  // ---- finalize + store ----
  float inv[8];
  #pragma unroll
  for (int r = 0; r < 8; ++r) inv[r] = 1.0f / lrow[r];
  #pragma unroll
  for (int nt = 0; nt < 8; ++nt) {
    #pragma unroll
    for (int r = 0; r < 8; ++r) {
      const int qrow = q0 + r + half * 8;
      const int dcol = nt * 16 + l16;
      Ob[(long)(b * SEQ + qrow) * DMODEL + h * HDIM + dcol] =
          f32_to_bf16(acc[nt].f[r] * inv[r]);
    }
  }
}

// ---------------------------------------------------------------------------
// Host launcher.  Inputs: x, start_pos, mask, wq, wk, wv, wo, cache_k, cache_v
// Workspace: Qb | Kb | Vt | AttnOut  (bf16, 32MB each = 128MB)
// ---------------------------------------------------------------------------
extern "C" void kernel_launch(void* const* d_in, const int* in_sizes, int n_in,
                              void* d_out, int out_size, void* d_ws, size_t ws_size,
                              hipStream_t stream) {
  const float* x  = (const float*)d_in[0];
  const float* wq = (const float*)d_in[3];
  const float* wk = (const float*)d_in[4];
  const float* wv = (const float*)d_in[5];
  const float* wo = (const float*)d_in[6];
  float* out = (float*)d_out;

  char* ws = (char*)d_ws;
  const size_t SZ = (size_t)MTOT * DMODEL * sizeof(bf16_t);  // 32MB
  bf16_t* Qb = (bf16_t*)(ws);
  bf16_t* Kb = (bf16_t*)(ws + SZ);
  bf16_t* Vt = (bf16_t*)(ws + 2 * SZ);
  bf16_t* Ab = (bf16_t*)(ws + 3 * SZ);

  dim3 grid(DMODEL / 256, MTOT / 128);   // 16 x 32
  dim3 blk(256);

  gemm_nt_wmma<1, 0><<<grid, blk, 0, stream>>>(x,  wq, Qb);   // Q
  gemm_nt_wmma<1, 0><<<grid, blk, 0, stream>>>(x,  wk, Kb);   // K
  gemm_nt_wmma<1, 1><<<grid, blk, 0, stream>>>(x,  wv, Vt);   // V (transposed store)
  flash_attn_wmma<<<1024, blk, 0, stream>>>(Qb, Kb, Vt, Ab);  // attention
  gemm_nt_wmma<0, 2><<<grid, blk, 0, stream>>>(Ab, wo, out);  // output projection
}
